// LSTM_ED_Attention_85057532330106
// MI455X (gfx1250) — matrix operand order repaired and checked by reading, not compiled, not run on hardware
//
#include <hip/hip_runtime.h>

// ---------------- types / helpers ----------------
typedef __attribute__((ext_vector_type(16))) __bf16 v16bf;
typedef __attribute__((ext_vector_type(8)))  float  v8f;

#define Bn   128
#define TEn  1024
#define TDn  128
#define INE  64
#define IND  32
#define Hn   256

__device__ __forceinline__ unsigned short f2bf(float x) {
    unsigned int u = __float_as_uint(x);
    unsigned int r = u + 0x7FFFu + ((u >> 16) & 1u);
    return (unsigned short)(r >> 16);
}
__device__ __forceinline__ float bf2f(unsigned short h) {
    return __uint_as_float(((unsigned int)h) << 16);
}
__device__ __forceinline__ float sigf(float x) { return 1.f / (1.f + expf(-x)); }

union ABits { uint4 q[2]; v16bf v; };

// ---------------- elementwise utility kernels ----------------
__global__ void f2bf_kernel(const float* __restrict__ in, unsigned short* __restrict__ out, long n) {
    long i = (long)blockIdx.x * 256 + threadIdx.x;
    if (i < n) out[i] = f2bf(in[i]);
}

__global__ void zero_kernel(unsigned int* __restrict__ p, long nwords) {
    long i = (long)blockIdx.x * 256 + threadIdx.x;
    if (i < nwords) p[i] = 0u;
}

__global__ void bias_add_kernel(const float* __restrict__ a, const float* __restrict__ b,
                                float* __restrict__ out, int n) {
    int i = blockIdx.x * 256 + threadIdx.x;
    if (i < n) out[i] = a[i] + b[i];
}

// Pack W rows (n,k) into WMMA B-fragment order (bf16), K split over two sources.
// element(n,k) = k<Ka ? Wa[n*aStride+aOff+k] : Wb[n*bStride+bOff+(k-Ka)]
__global__ void pack_kernel(const float* __restrict__ Wa, int aStride, int aOff, int Ka,
                            const float* __restrict__ Wb, int bStride, int bOff, int Kb,
                            int N, unsigned short* __restrict__ out) {
    int Ktot = Ka + Kb;
    long i = (long)blockIdx.x * 256 + threadIdx.x;
    if (i >= (long)N * Ktot) return;
    int n = (int)(i / Ktot);
    int k = (int)(i - (long)n * Ktot);
    float v = (k < Ka) ? Wa[(long)n * aStride + aOff + k]
                       : Wb[(long)n * bStride + bOff + (k - Ka)];
    // B-fragment order used here (A uses the same): lanes 0-15 hold K 0-15, lanes 16-31 hold K 16-31 (pairs per VGPR)
    int kt   = k >> 5;
    int ntl  = n >> 4;
    int lane = (n & 15) | (((k >> 4) & 1) << 4);
    int e    = k & 15;
    int Ntiles = N >> 4;
    out[(((long)kt * Ntiles + ntl) * 32 + lane) * 16 + e] = f2bf(v);
}

// enc_out[t][b][h] = 0.5*(ycat1[t][b][h] + ycat1[t][b][H+h])  (bf16)
__global__ void encout_kernel(const unsigned short* __restrict__ ycat1,
                              unsigned short* __restrict__ eo, long n) {
    long i = (long)blockIdx.x * 256 + threadIdx.x;
    if (i >= n) return;
    long row = i >> 8; int h = (int)(i & 255);
    eo[i] = f2bf(0.5f * (bf2f(ycat1[row * 512 + h]) + bf2f(ycat1[row * 512 + 256 + h])));
}

// decoder init states: l0 = avg of fwd states (enc l0,l1); l1 = avg of bwd states
__global__ void decinit_kernel(const unsigned short* h0f, const unsigned short* h0b,
                               const unsigned short* h1f, const unsigned short* h1b,
                               const float* c0f, const float* c0b,
                               const float* c1f, const float* c1b,
                               unsigned short* dh0, unsigned short* dh1,
                               float* dc0, float* dc1) {
    int i = blockIdx.x * 256 + threadIdx.x;
    if (i >= Bn * Hn) return;
    dh0[i] = f2bf(0.5f * (bf2f(h0f[i]) + bf2f(h1f[i])));
    dh1[i] = f2bf(0.5f * (bf2f(h0b[i]) + bf2f(h1b[i])));
    dc0[i] = 0.5f * (c0f[i] + c1f[i]);
    dc1[i] = 0.5f * (c0b[i] + c1b[i]);
}

// ---------------- fused LSTM step: gates GEMM (WMMA bf16) + cell ----------------
// Grid (8,8): blockIdx.x = batch tile (16 rows), blockIdx.y = 32-wide hidden chunk.
// X = [src0 | src1 | hprev], W packed as frags over N=4H=1024, Ktot=in0+in1+H.
// LDS A panel row stride padded by 8 halves (16B) to keep b128 alignment + spread banks.
__global__ __launch_bounds__(256) void lstm_step_kernel(
    const unsigned short* __restrict__ s0, long s0s, int in0,
    const unsigned short* __restrict__ s1, long s1s, int in1,
    const unsigned short* __restrict__ hp, long hps,
    const unsigned short* __restrict__ wpk,
    const float* __restrict__ bias,
    float* __restrict__ cst,
    unsigned short* __restrict__ hst,
    unsigned short* __restrict__ yout, long ys)
{
    const int Ktot = in0 + in1 + Hn;           // <= 768, multiple of 32
    const int KP   = Ktot + 8;                 // padded LDS row stride (halves)
    __shared__ unsigned short shA[16 * (768 + 8)];
    __shared__ float shG[8 * 256];
    const int tid = threadIdx.x;
    const int m0  = blockIdx.x * 16;
    const int jc  = blockIdx.y;                // 0..7

    for (int i = tid; i < 16 * Ktot; i += 256) {
        int row = i / Ktot; int k = i - row * Ktot;
        int m = m0 + row;
        unsigned short v;
        if (k < in0)              v = s0[(long)m * s0s + k];
        else if (k < in0 + in1)   v = s1[(long)m * s1s + (k - in0)];
        else                      v = hp[(long)m * hps + (k - in0 - in1)];
        shA[row * KP + k] = v;
    }
    __syncthreads();

    const int w = tid >> 5, lane = tid & 31;
    const int gate = w >> 1, jt = w & 1;
    const int n0 = gate * Hn + jc * 32 + jt * 16;
    const int nt = n0 >> 4;
    const int Ntiles = (4 * Hn) >> 4;          // 64
    const uint4* shA128 = (const uint4*)shA;
    const int arow = lane & 15;
    const int koff = (lane >> 4) * 16;
    const int abase = arow * KP + koff;        // halves; multiple of 8
    const v16bf* Bp = (const v16bf*)wpk;

    v8f acc = {};
    const int nkt = Ktot >> 5;
    for (int kt = 0; kt < nkt; ++kt) {
        int k0 = kt * 32;
        ABits a;
        a.q[0] = shA128[(abase + k0) >> 3];        // halves k: koff+0..7
        a.q[1] = shA128[(abase + k0 + 8) >> 3];    // halves k: koff+8..15
        v16bf bv = Bp[((long)kt * Ntiles + nt) * 32 + lane];
        acc = __builtin_amdgcn_wmma_f32_16x16x32_bf16(false, a.v, false, bv,
                                                      (short)0, acc, false, false);
    }

    const int nn = lane & 15;
    float bv = bias[n0 + nn];
#pragma unroll
    for (int r = 0; r < 8; ++r) {
        int m = ((lane >> 4) << 3) + r;
        shG[w * 256 + m * 16 + nn] = acc[r] + bv;
    }
    __syncthreads();

    for (int i = tid; i < 512; i += 256) {
        int m = i >> 5, jj = i & 31;
        int hi = jj >> 4, lo = jj & 15;
        float ig = shG[(0 + hi) * 256 + m * 16 + lo];
        float fg = shG[(2 + hi) * 256 + m * 16 + lo];
        float gg = shG[(4 + hi) * 256 + m * 16 + lo];
        float og = shG[(6 + hi) * 256 + m * 16 + lo];
        int b = m0 + m;
        int j = jc * 32 + jj;
        long idx = (long)b * Hn + j;
        float c = sigf(fg) * cst[idx] + sigf(ig) * tanhf(gg);
        float h = sigf(og) * tanhf(c);
        cst[idx] = c;
        unsigned short hb = f2bf(h);
        if (hst)  hst[idx] = hb;
        if (yout) yout[(long)b * ys + j] = hb;
    }
}

// ---------------- generic WMMA GEMM: out[M][N] = A[M][K](bf16) * Wpk + bias ----------------
// Grid (M/16, N/128), 8 waves/block, one 16x16 tile per wave. K <= 256.
__global__ __launch_bounds__(256) void gemm_bias_kernel(
    const unsigned short* __restrict__ A, int K,
    const unsigned short* __restrict__ wpk, int N,
    const float* __restrict__ bias,
    float* __restrict__ out)
{
    const int KP = K + 8;
    __shared__ unsigned short shA[16 * (256 + 8)];
    const int tid = threadIdx.x;
    const long m0 = (long)blockIdx.x * 16;
    for (int i = tid; i < 16 * K; i += 256) {
        int row = i / K, k = i - row * K;
        shA[row * KP + k] = A[(m0 + row) * K + k];
    }
    __syncthreads();

    const int w = tid >> 5, lane = tid & 31;
    const int n0 = blockIdx.y * 128 + w * 16;
    const int nt = n0 >> 4, Ntiles = N >> 4;
    const uint4* shA128 = (const uint4*)shA;
    const int arow = lane & 15;
    const int koff = (lane >> 4) * 16;
    const int abase = arow * KP + koff;
    const v16bf* Bp = (const v16bf*)wpk;

    v8f acc = {};
    for (int kt = 0; kt < (K >> 5); ++kt) {
        int k0 = kt * 32;
        ABits a;
        a.q[0] = shA128[(abase + k0) >> 3];
        a.q[1] = shA128[(abase + k0 + 8) >> 3];
        v16bf bv = Bp[((long)kt * Ntiles + nt) * 32 + lane];
        acc = __builtin_amdgcn_wmma_f32_16x16x32_bf16(false, a.v, false, bv,
                                                      (short)0, acc, false, false);
    }
    const int nn = lane & 15;
    float bv = bias ? bias[n0 + nn] : 0.f;
#pragma unroll
    for (int r = 0; r < 8; ++r) {
        long m = m0 + ((lane >> 4) << 3) + r;
        out[m * N + n0 + nn] = acc[r] + bv;
    }
}

// ---------------- attention scores: e[b][t] = sum_k tanh(q[b][k]+ep[t*B+b][k]) * v[k] ----------------
__global__ __launch_bounds__(256) void scores_kernel(
    const float* __restrict__ q, const float* __restrict__ ep,
    const float* __restrict__ v, float* __restrict__ sc)
{
    int gw = (blockIdx.x * 256 + threadIdx.x) >> 5;
    int lane = threadIdx.x & 31;
    int b = gw >> 10;            // / TE
    int t = gw & 1023;
    const float* qb = q + (long)b * 256;
    const float* er = ep + ((long)t * Bn + b) * 256;
    float s = 0.f;
#pragma unroll
    for (int j = 0; j < 8; ++j) {
        int k = lane * 8 + j;
        s += tanhf(qb[k] + er[k]) * v[k];
    }
    for (int off = 16; off; off >>= 1) s += __shfl_xor(s, off, 32);
    if (lane == 0) sc[(long)b * TEn + t] = s;
}

// ---------------- softmax over TE + context: ctx[b][h] (bf16) ----------------
__global__ __launch_bounds__(256) void softmax_ctx_kernel(
    const float* __restrict__ sc, const unsigned short* __restrict__ eo,
    unsigned short* __restrict__ ctx)
{
    __shared__ float shW[TEn];
    __shared__ float red[256];
    int b = blockIdx.x, tid = threadIdx.x;
    const float* s = sc + (long)b * TEn;
    float mx = -1e30f;
    for (int t = tid; t < TEn; t += 256) mx = fmaxf(mx, s[t]);
    red[tid] = mx; __syncthreads();
    for (int o = 128; o; o >>= 1) { if (tid < o) red[tid] = fmaxf(red[tid], red[tid + o]); __syncthreads(); }
    mx = red[0]; __syncthreads();
    float sum = 0.f;
    for (int t = tid; t < TEn; t += 256) { float e = expf(s[t] - mx); shW[t] = e; sum += e; }
    red[tid] = sum; __syncthreads();
    for (int o = 128; o; o >>= 1) { if (tid < o) red[tid] += red[tid + o]; __syncthreads(); }
    float inv = 1.f / red[0];
    int h = tid;
    float acc = 0.f;
    for (int t = 0; t < TEn; ++t)
        acc += shW[t] * bf2f(eo[((long)t * Bn + b) * 256 + h]);
    ctx[(long)b * 256 + h] = f2bf(acc * inv);
}

// ---------------- final FC: out[b][t] = ydec[t][b][:] . fcW + fcb ----------------
__global__ __launch_bounds__(256) void fc_kernel(
    const unsigned short* __restrict__ ydec, const float* __restrict__ Wf,
    const float* __restrict__ bf_, float* __restrict__ out)
{
    int gw = (blockIdx.x * 256 + threadIdx.x) >> 5;
    int lane = threadIdx.x & 31;
    int b = gw >> 7;             // / TD
    int t = gw & 127;
    const unsigned short* yr = ydec + ((long)t * Bn + b) * 256;
    float s = 0.f;
#pragma unroll
    for (int j = 0; j < 8; ++j) {
        int h = lane + 32 * j;
        s += bf2f(yr[h]) * Wf[h];
    }
    for (int off = 16; off; off >>= 1) s += __shfl_xor(s, off, 32);
    if (lane == 0) out[(long)b * TDn + t] = s + bf_[0];
}

// ---------------- host ----------------
static inline long cdiv(long a, long b) { return (a + b - 1) / b; }

extern "C" void kernel_launch(void* const* d_in, const int* in_sizes, int n_in,
                              void* d_out, int out_size, void* d_ws, size_t ws_size,
                              hipStream_t stream) {
    (void)in_sizes; (void)n_in; (void)out_size; (void)ws_size;
    const float* x_e = (const float*)d_in[0];
    const float* x_d = (const float*)d_in[1];
    const float* eWih[4]; const float* eWhh[4]; const float* ebih[4]; const float* ebhh[4];
    for (int i = 0; i < 4; ++i) {
        eWih[i] = (const float*)d_in[2 + i];
        eWhh[i] = (const float*)d_in[6 + i];
        ebih[i] = (const float*)d_in[10 + i];
        ebhh[i] = (const float*)d_in[14 + i];
    }
    const float* dWih[2]; const float* dWhh[2]; const float* dbih[2]; const float* dbhh[2];
    for (int i = 0; i < 2; ++i) {
        dWih[i] = (const float*)d_in[18 + i];
        dWhh[i] = (const float*)d_in[20 + i];
        dbih[i] = (const float*)d_in[22 + i];
        dbhh[i] = (const float*)d_in[24 + i];
    }
    const float* attnW = (const float*)d_in[26];
    const float* attnB = (const float*)d_in[27];
    const float* attnV = (const float*)d_in[28];
    const float* fcW   = (const float*)d_in[29];
    const float* fcB   = (const float*)d_in[30];
    float* outp = (float*)d_out;

    // workspace layout
    char* ws = (char*)d_ws;
    size_t off = 0;
    auto alloc = [&](size_t bytes) -> void* {
        void* p = ws + off;
        off += (bytes + 255) & ~(size_t)255;
        return p;
    };
    unsigned short* xeb = (unsigned short*)alloc((size_t)Bn * TEn * INE * 2);
    unsigned short* xdb = (unsigned short*)alloc((size_t)Bn * TDn * IND * 2);
    const int encK[2]  = {INE + Hn, 2 * Hn + Hn};            // 320, 768
    unsigned short* wpkE[4];
    for (int i = 0; i < 4; ++i) wpkE[i] = (unsigned short*)alloc((size_t)1024 * encK[i / 2] * 2);
    const int decK[2]  = {IND + Hn + Hn, Hn + Hn};           // 544, 512
    unsigned short* wpkD[2];
    for (int i = 0; i < 2; ++i) wpkD[i] = (unsigned short*)alloc((size_t)1024 * decK[i] * 2);
    unsigned short* wpkWh = (unsigned short*)alloc((size_t)256 * 256 * 2);
    unsigned short* wpkWe = (unsigned short*)alloc((size_t)256 * 256 * 2);
    float* biasE[4]; for (int i = 0; i < 4; ++i) biasE[i] = (float*)alloc(1024 * 4);
    float* biasD[2]; for (int i = 0; i < 2; ++i) biasD[i] = (float*)alloc(1024 * 4);
    unsigned short* ycat0 = (unsigned short*)alloc((size_t)TEn * Bn * 512 * 2);
    unsigned short* ycat1 = (unsigned short*)alloc((size_t)TEn * Bn * 512 * 2);
    unsigned short* eo    = (unsigned short*)alloc((size_t)TEn * Bn * 256 * 2);
    float* eproj          = (float*)alloc((size_t)TEn * Bn * 256 * 4);
    // contiguous zero region: zbuf + enc h states (bf16) + enc c states (f32)
    size_t zbytes = (size_t)Bn * Hn * 2 * 5 + (size_t)Bn * Hn * 4 * 4;
    char* zreg = (char*)alloc(zbytes);
    unsigned short* zbuf = (unsigned short*)zreg;
    unsigned short* ehs[4]; float* ecs[4];
    for (int i = 0; i < 4; ++i) ehs[i] = (unsigned short*)(zreg + (size_t)Bn * Hn * 2 * (1 + i));
    for (int i = 0; i < 4; ++i) ecs[i] = (float*)(zreg + (size_t)Bn * Hn * 2 * 5 + (size_t)Bn * Hn * 4 * i);
    unsigned short* dh0 = (unsigned short*)alloc((size_t)Bn * Hn * 2);
    unsigned short* dh1 = (unsigned short*)alloc((size_t)Bn * Hn * 2);
    float* dc0 = (float*)alloc((size_t)Bn * Hn * 4);
    float* dc1 = (float*)alloc((size_t)Bn * Hn * 4);
    unsigned short* hl0[2];
    hl0[0] = (unsigned short*)alloc((size_t)Bn * Hn * 2);
    hl0[1] = (unsigned short*)alloc((size_t)Bn * Hn * 2);
    float* qb  = (float*)alloc((size_t)Bn * 256 * 4);
    float* scb = (float*)alloc((size_t)Bn * TEn * 4);
    unsigned short* ctx = (unsigned short*)alloc((size_t)Bn * 256 * 2);
    unsigned short* ydec = (unsigned short*)alloc((size_t)TDn * Bn * 256 * 2);

    // --- preprocessing ---
    {
        long n = (long)Bn * TEn * INE;
        f2bf_kernel<<<cdiv(n, 256), 256, 0, stream>>>(x_e, xeb, n);
        n = (long)Bn * TDn * IND;
        f2bf_kernel<<<cdiv(n, 256), 256, 0, stream>>>(x_d, xdb, n);
        zero_kernel<<<cdiv((long)(zbytes / 4), 256), 256, 0, stream>>>((unsigned int*)zreg, (long)(zbytes / 4));
    }
    for (int i = 0; i < 4; ++i) {
        int Ka = (i < 2) ? INE : 2 * Hn;
        long n = (long)1024 * (Ka + Hn);
        pack_kernel<<<cdiv(n, 256), 256, 0, stream>>>(eWih[i], Ka, 0, Ka, eWhh[i], Hn, 0, Hn, 1024, wpkE[i]);
        bias_add_kernel<<<4, 256, 0, stream>>>(ebih[i], ebhh[i], biasE[i], 1024);
    }
    {
        int Ka0 = IND + Hn;  // 288
        long n = (long)1024 * decK[0];
        pack_kernel<<<cdiv(n, 256), 256, 0, stream>>>(dWih[0], Ka0, 0, Ka0, dWhh[0], Hn, 0, Hn, 1024, wpkD[0]);
        n = (long)1024 * decK[1];
        pack_kernel<<<cdiv(n, 256), 256, 0, stream>>>(dWih[1], Hn, 0, Hn, dWhh[1], Hn, 0, Hn, 1024, wpkD[1]);
        bias_add_kernel<<<4, 256, 0, stream>>>(dbih[0], dbhh[0], biasD[0], 1024);
        bias_add_kernel<<<4, 256, 0, stream>>>(dbih[1], dbhh[1], biasD[1], 1024);
        long na = (long)256 * 256;
        pack_kernel<<<cdiv(na, 256), 256, 0, stream>>>(attnW, 512, 0, 256, nullptr, 0, 0, 0, 256, wpkWh);
        pack_kernel<<<cdiv(na, 256), 256, 0, stream>>>(attnW, 512, 256, 256, nullptr, 0, 0, 0, 256, wpkWe);
    }

    // --- encoder: 2 layers x bidirectional, fused step kernels ---
    dim3 sg(8, 8);
    for (int l = 0; l < 2; ++l) {
        unsigned short* ycl = (l == 0) ? ycat0 : ycat1;
        int in0 = (l == 0) ? INE : 512;
        long s0s = (l == 0) ? (long)TEn * INE : 512;
        for (int t = 0; t < TEn; ++t) {
            // forward (state index 2l)
            const unsigned short* s0f = (l == 0) ? (xeb + (long)t * INE) : (ycat0 + (long)t * Bn * 512);
            const unsigned short* hpf = (t == 0) ? zbuf : (ycl + (long)(t - 1) * Bn * 512);
            long hpsf = (t == 0) ? Hn : 512;
            lstm_step_kernel<<<sg, 256, 0, stream>>>(s0f, s0s, in0, nullptr, 0, 0,
                hpf, hpsf, wpkE[2 * l], biasE[2 * l], ecs[2 * l], ehs[2 * l],
                ycl + (long)t * Bn * 512, 512);
            // backward (state index 2l+1), processes time tr = TE-1-t
            int tr = TEn - 1 - t;
            const unsigned short* s0b = (l == 0) ? (xeb + (long)tr * INE) : (ycat0 + (long)tr * Bn * 512);
            const unsigned short* hpb = (t == 0) ? zbuf : (ycl + (long)(tr + 1) * Bn * 512 + 256);
            long hpsb = (t == 0) ? Hn : 512;
            lstm_step_kernel<<<sg, 256, 0, stream>>>(s0b, s0s, in0, nullptr, 0, 0,
                hpb, hpsb, wpkE[2 * l + 1], biasE[2 * l + 1], ecs[2 * l + 1], ehs[2 * l + 1],
                ycl + (long)tr * Bn * 512 + 256, 512);
        }
    }

    // enc_out + enc_proj (attn_b folded as GEMM bias)
    {
        long n = (long)TEn * Bn * 256;
        encout_kernel<<<cdiv(n, 256), 256, 0, stream>>>(ycat1, eo, n);
        dim3 g((TEn * Bn) / 16, 2);
        gemm_bias_kernel<<<g, 256, 0, stream>>>(eo, 256, wpkWe, 256, attnB, eproj);
        decinit_kernel<<<cdiv((long)Bn * Hn, 256), 256, 0, stream>>>(
            ehs[0], ehs[1], ehs[2], ehs[3], ecs[0], ecs[1], ecs[2], ecs[3],
            dh0, dh1, dc0, dc1);
    }

    // --- decoder: attention + 2 LSTM layers per step ---
    for (int t = 0; t < TDn; ++t) {
        const unsigned short* hid = (t == 0) ? dh1 : (ydec + (long)(t - 1) * Bn * 256);
        gemm_bias_kernel<<<dim3(Bn / 16, 2), 256, 0, stream>>>(hid, 256, wpkWh, 256, nullptr, qb);
        scores_kernel<<<(Bn * TEn) / 8, 256, 0, stream>>>(qb, eproj, attnV, scb);
        softmax_ctx_kernel<<<Bn, 256, 0, stream>>>(scb, eo, ctx);
        const unsigned short* h0p = (t == 0) ? dh0 : hl0[(t - 1) & 1];
        lstm_step_kernel<<<sg, 256, 0, stream>>>(
            xdb + (long)t * IND, (long)TDn * IND, IND, ctx, 256, 256,
            h0p, 256, wpkD[0], biasD[0], dc0, hl0[t & 1], nullptr, 0);
        const unsigned short* h1p = (t == 0) ? dh1 : (ydec + (long)(t - 1) * Bn * 256);
        lstm_step_kernel<<<sg, 256, 0, stream>>>(
            hl0[t & 1], 256, 256, nullptr, 0, 0,
            h1p, 256, wpkD[1], biasD[1], dc1, nullptr,
            ydec + (long)t * Bn * 256, 256);
    }

    // final FC
    fc_kernel<<<(Bn * TDn) / 8, 256, 0, stream>>>(ydec, fcW, fcB, outp);
}
